// RNN_13597866459611
// MI455X (gfx1250) — matrix operand order repaired and checked by reading, not compiled, not run on hardware
//
#include <hip/hip_runtime.h>
#include <math.h>

typedef __bf16 bf16t;
typedef __attribute__((ext_vector_type(16))) __bf16 v16bf;
typedef __attribute__((ext_vector_type(8)))  float v8f;
typedef __attribute__((ext_vector_type(4)))  float v4f;

#define NW_   5000
#define LC_   12
#define LW_   32
#define NSENT_ 192
#define DH_   256

// ---------------------------------------------------------------------------
// WMMA helpers (CDNA5 wave32, v_wmma_f32_16x16x32_bf16)
// A layout (16x32 bf16): lane<16 -> M=lane, elems 0..7 = K base+0..7, 8..15 = K base+16..23
//                        lane>=16 -> M=lane-16, K offset +8 (i.e. 8..15 / 24..31)
// B layout (32x16 bf16): lane&15 = N, elems e: K = base + (lane&16?8:0) + (e&7) + (e&8?16:0)
// C/D layout: lane&15 = N, vgpr i -> M = i + (lane&16 ? 8 : 0)
// ---------------------------------------------------------------------------

__device__ __forceinline__ v8f wmma_bf16(v16bf a, v16bf b, v8f c) {
  return __builtin_amdgcn_wmma_f32_16x16x32_bf16(false, a, false, b, (short)0, c,
                                                 false, false);
}

// Pack one A fragment from a row-major f32 row (row must be 16B aligned,
// kbase multiple of 32; loads row[kbase + off .. +7] and [+16 .. +23]).
__device__ __forceinline__ v16bf packA(const float* __restrict__ row, int kbase, int lane) {
  const float* p = row + kbase + ((lane & 16) ? 8 : 0);
  v4f x0 = *(const v4f*)(p + 0);
  v4f x1 = *(const v4f*)(p + 4);
  v4f x2 = *(const v4f*)(p + 16);
  v4f x3 = *(const v4f*)(p + 20);
  v16bf a;
#pragma unroll
  for (int i = 0; i < 4; ++i) {
    a[i]      = (__bf16)x0[i];
    a[4 + i]  = (__bf16)x1[i];
    a[8 + i]  = (__bf16)x2[i];
    a[12 + i] = (__bf16)x3[i];
  }
  return a;
}

// Pre-packed B fragment: contiguous 32B per (nTile,kTile,lane).
__device__ __forceinline__ v16bf loadB(const bf16t* __restrict__ Bp, int KT, int nt,
                                       int kt, int lane) {
  const bf16t* p = Bp + ((((size_t)nt * KT + kt) * 32 + (size_t)lane) << 4);
  return *(const v16bf*)p;
}

// ---------------------------------------------------------------------------
// Weight packing: W row-major (K x N) f32 -> bf16 fragments, K zero-padded.
// ---------------------------------------------------------------------------
__global__ __launch_bounds__(256)
void pack_w_k(const float* __restrict__ W, int rowOff, int Ksrc, int KT, int N,
              bf16t* __restrict__ Bp) {
  size_t idx = (size_t)blockIdx.x * 256 + threadIdx.x;
  size_t total = (size_t)(N / 16) * KT * 512;
  if (idx >= total) return;
  int e    = (int)(idx & 15);
  size_t t1 = idx >> 4;
  int lane = (int)(t1 & 31);
  size_t t2 = t1 >> 5;
  int kt = (int)(t2 % KT);
  int nt = (int)(t2 / KT);
  int n = nt * 16 + (lane & 15);
  int k = kt * 32 + ((lane & 16) ? 8 : 0) + (e & 7) + ((e & 8) ? 16 : 0);
  Bp[idx] = (k < Ksrc) ? (__bf16)W[(size_t)(rowOff + k) * N + n] : (__bf16)0.0f;
}

__global__ __launch_bounds__(256)
void zero_k(float* __restrict__ p, size_t n) {
  size_t i = (size_t)blockIdx.x * 256 + threadIdx.x;
  if (i < n) p[i] = 0.0f;
}

// X_char: (NW*LC) x 128, cols 100..127 zero
__global__ __launch_bounds__(256)
void build_xchar_k(const int* __restrict__ w2c, const float* __restrict__ cemb,
                   float* __restrict__ X) {
  size_t idx = (size_t)blockIdx.x * 256 + threadIdx.x;
  size_t total = (size_t)NW_ * LC_ * 128;
  if (idx >= total) return;
  int k = (int)(idx & 127);
  size_t r = idx >> 7;                // word*LC + t
  float v = 0.0f;
  if (k < 100) {
    int c = w2c[r];
    v = cemb[(size_t)c * 100 + k];
  }
  X[idx] = v;
}

// inputs_s: (NSENT*LW) x 640; cols 0..511 char_states gather, 512..611 word_emb, rest 0
__global__ __launch_bounds__(256)
void build_inputs_k(const int* __restrict__ s2w, const int* __restrict__ b2s,
                    const float* __restrict__ CS, const float* __restrict__ wemb,
                    float* __restrict__ X) {
  size_t idx = (size_t)blockIdx.x * 256 + threadIdx.x;
  size_t total = (size_t)NSENT_ * LW_ * 640;
  if (idx >= total) return;
  int k = (int)(idx % 640);
  size_t r = idx / 640;
  int i = (int)(r / LW_);
  int t = (int)(r % LW_);
  float v = 0.0f;
  if (k < 512) {
    int wc = s2w[(size_t)i * LW_ + t];          // raw sentence order (reference)
    v = CS[(size_t)wc * 512 + k];
  } else if (k < 612) {
    int sidv = b2s[i];
    int ww = s2w[(size_t)sidv * LW_ + t];       // b2s-mapped order (reference)
    v = wemb[(size_t)ww * 100 + (k - 512)];
  }
  X[idx] = v;
}

// ---------------------------------------------------------------------------
// GRU gate step: G = sigmoid(x_t @ gk_x + H @ gk_h + gb); write RH = r*H, U = u
// Wave computes a 16(M) x 64(N) tile; grid = (ceil(M/16), 512/64)
// ---------------------------------------------------------------------------
__global__ __launch_bounds__(32)
void gru_gate_k(const float* __restrict__ X, int lda0, int KTx, int T,
                const int* __restrict__ lens, const int* __restrict__ sid,
                int t, int dir,
                const float* __restrict__ H, int ldh,
                const bf16t* __restrict__ Bx, const bf16t* __restrict__ Bh,
                const float* __restrict__ gbias,
                float* __restrict__ RH, float* __restrict__ U, int M) {
  const int lane = threadIdx.x;
  const int tm = blockIdx.x;
  const int ng = blockIdx.y;
  int mA = tm * 16 + (lane & 15);
  int mc = mA < M ? mA : M - 1;
  int li = sid ? sid[mc] : mc;
  int l = lens[li]; if (l < 1) l = 1;
  int tsel = dir ? (l - 1 - t) : t;
  if (tsel < 0) tsel = 0;
  if (tsel > T - 1) tsel = T - 1;
  const float* a0 = X + (size_t)(mc * T + tsel) * lda0;
  const float* a1 = H + (size_t)mc * ldh;

  v8f acc[4] = {};
  for (int kt = 0; kt < KTx; ++kt) {
    v16bf a = packA(a0, kt * 32, lane);
#pragma unroll
    for (int j = 0; j < 4; ++j)
      acc[j] = wmma_bf16(a, loadB(Bx, KTx, ng * 4 + j, kt, lane), acc[j]);
  }
#pragma unroll
  for (int kt = 0; kt < 8; ++kt) {
    v16bf a = packA(a1, kt * 32, lane);
#pragma unroll
    for (int j = 0; j < 4; ++j)
      acc[j] = wmma_bf16(a, loadB(Bh, 8, ng * 4 + j, kt, lane), acc[j]);
  }

  int mBase = tm * 16 + ((lane & 16) ? 8 : 0);
#pragma unroll
  for (int j = 0; j < 4; ++j) {
    int n = ng * 64 + j * 16 + (lane & 15);
    float bb = gbias[n];
#pragma unroll
    for (int i = 0; i < 8; ++i) {
      int mm = mBase + i;
      if (mm < M) {
        float g = 1.0f / (1.0f + __expf(-(acc[j][i] + bb)));
        if (n < DH_) RH[(size_t)mm * DH_ + n] = g * H[(size_t)mm * ldh + n];
        else         U[(size_t)mm * DH_ + (n - DH_)] = g;
      }
    }
  }
}

// ---------------------------------------------------------------------------
// GRU update step: C = tanh(x_t @ ck_x + RH @ ck_h + cb); H = mask ? u*H+(1-u)*C : H
// grid = (ceil(M/16), 256/64)
// ---------------------------------------------------------------------------
__global__ __launch_bounds__(32)
void gru_update_k(const float* __restrict__ X, int lda0, int KTx, int T,
                  const int* __restrict__ lens, const int* __restrict__ sid,
                  int t, int dir,
                  float* __restrict__ H, int ldh,
                  const float* __restrict__ RH,
                  const bf16t* __restrict__ Bx, const bf16t* __restrict__ Bh,
                  const float* __restrict__ cbias,
                  const float* __restrict__ U, int M) {
  const int lane = threadIdx.x;
  const int tm = blockIdx.x;
  const int ng = blockIdx.y;
  int mA = tm * 16 + (lane & 15);
  int mc = mA < M ? mA : M - 1;
  int liA = sid ? sid[mc] : mc;
  int lA = lens[liA]; if (lA < 1) lA = 1;
  int tsel = dir ? (lA - 1 - t) : t;
  if (tsel < 0) tsel = 0;
  if (tsel > T - 1) tsel = T - 1;
  const float* a0 = X + (size_t)(mc * T + tsel) * lda0;
  const float* a1 = RH + (size_t)mc * DH_;

  v8f acc[4] = {};
  for (int kt = 0; kt < KTx; ++kt) {
    v16bf a = packA(a0, kt * 32, lane);
#pragma unroll
    for (int j = 0; j < 4; ++j)
      acc[j] = wmma_bf16(a, loadB(Bx, KTx, ng * 4 + j, kt, lane), acc[j]);
  }
#pragma unroll
  for (int kt = 0; kt < 8; ++kt) {
    v16bf a = packA(a1, kt * 32, lane);
#pragma unroll
    for (int j = 0; j < 4; ++j)
      acc[j] = wmma_bf16(a, loadB(Bh, 8, ng * 4 + j, kt, lane), acc[j]);
  }

  int mBase = tm * 16 + ((lane & 16) ? 8 : 0);
#pragma unroll
  for (int j = 0; j < 4; ++j) {
    int n = ng * 64 + j * 16 + (lane & 15);
    float bb = cbias[n];
#pragma unroll
    for (int i = 0; i < 8; ++i) {
      int mm = mBase + i;
      if (mm < M) {
        float c = tanhf(acc[j][i] + bb);
        float h = H[(size_t)mm * ldh + n];
        float u = U[(size_t)mm * DH_ + n];
        int li = sid ? sid[mm] : mm;
        int l = lens[li]; if (l < 1) l = 1;
        float hn = u * h + (1.0f - u) * c;
        H[(size_t)mm * ldh + n] = (t < l) ? hn : h;
      }
    }
  }
}

// ---------------------------------------------------------------------------
// MLP GEMM: Out = act(A @ Bp + bias); act=1 -> leaky ReLU(0.2)
// grid = (M/16, N/64)
// ---------------------------------------------------------------------------
__global__ __launch_bounds__(32)
void mlp_gemm_k(const float* __restrict__ A, int lda,
                const bf16t* __restrict__ Bp, int KT,
                const float* __restrict__ bias,
                float* __restrict__ Out, int ldo, int M, int act) {
  const int lane = threadIdx.x;
  const int tm = blockIdx.x;
  const int ng = blockIdx.y;
  int mA = tm * 16 + (lane & 15);
  int mc = mA < M ? mA : M - 1;
  const float* arow = A + (size_t)mc * lda;

  v8f acc[4] = {};
  for (int kt = 0; kt < KT; ++kt) {
    v16bf a = packA(arow, kt * 32, lane);
#pragma unroll
    for (int j = 0; j < 4; ++j)
      acc[j] = wmma_bf16(a, loadB(Bp, KT, ng * 4 + j, kt, lane), acc[j]);
  }

  int mBase = tm * 16 + ((lane & 16) ? 8 : 0);
#pragma unroll
  for (int j = 0; j < 4; ++j) {
    int n = ng * 64 + j * 16 + (lane & 15);
    float bb = bias[n];
#pragma unroll
    for (int i = 0; i < 8; ++i) {
      int mm = mBase + i;
      if (mm < M) {
        float v = acc[j][i] + bb;
        if (act) v = (v > 0.0f) ? v : 0.2f * v;
        Out[(size_t)mm * ldo + n] = v;
      }
    }
  }
}

// out(32x2) = fc(32x512) @ ow(512x2) + ob
__global__ __launch_bounds__(64)
void final_out_k(const float* __restrict__ fc, const float* __restrict__ ow,
                 const float* __restrict__ ob, float* __restrict__ out) {
  int tid = threadIdx.x;
  int b = tid >> 1, c = tid & 1;
  float s = ob[c];
  for (int k = 0; k < 512; ++k) s += fc[(size_t)b * 512 + k] * ow[(size_t)k * 2 + c];
  out[(size_t)b * 2 + c] = s;
}

// ---------------------------------------------------------------------------
extern "C" void kernel_launch(void* const* d_in, const int* in_sizes, int n_in,
                              void* d_out, int out_size, void* d_ws, size_t ws_size,
                              hipStream_t stream) {
  (void)in_sizes; (void)n_in; (void)out_size;
  const int*   w2c  = (const int*)d_in[0];
  const int*   wlen = (const int*)d_in[1];
  const int*   s2w  = (const int*)d_in[2];
  const int*   slen = (const int*)d_in[3];
  const int*   b2s  = (const int*)d_in[4];
  const float* cemb = (const float*)d_in[5];
  const float* wemb = (const float*)d_in[6];
  const float* c_gk = (const float*)d_in[7];
  const float* c_gb = (const float*)d_in[8];
  const float* c_ck = (const float*)d_in[9];
  const float* c_cb = (const float*)d_in[10];
  const float* s_gk = (const float*)d_in[11];
  const float* s_gb = (const float*)d_in[12];
  const float* s_ck = (const float*)d_in[13];
  const float* s_cb = (const float*)d_in[14];
  const float* sw1  = (const float*)d_in[15];
  const float* sb1  = (const float*)d_in[16];
  const float* sw2  = (const float*)d_in[17];
  const float* sb2  = (const float*)d_in[18];
  const float* ew1  = (const float*)d_in[19];
  const float* eb1  = (const float*)d_in[20];
  const float* ew2  = (const float*)d_in[21];
  const float* eb2  = (const float*)d_in[22];
  const float* cw1  = (const float*)d_in[23];
  const float* cb1  = (const float*)d_in[24];
  const float* cw2  = (const float*)d_in[25];
  const float* cb2  = (const float*)d_in[26];
  const float* ow   = (const float*)d_in[27];
  const float* obv  = (const float*)d_in[28];
  float* out = (float*)d_out;

  // -------- workspace layout (f32 region then bf16 packed-weight region) ----
  float* wsf = (float*)d_ws;
  size_t of = 0;
  auto afl = [&](size_t n) { size_t r = of; of += (n + 63) & ~(size_t)63; return r; };
  size_t oX  = afl((size_t)NW_ * LC_ * 128);   // char inputs, K-padded
  size_t oIn = afl((size_t)NSENT_ * LW_ * 640); // sentence inputs, K-padded
  size_t oCS = afl((size_t)NW_ * 512);          // char_states [h_fw | h_bw]
  size_t oSS = afl((size_t)NSENT_ * 512);       // sent_states [h_fw | h_bw]
  size_t oRH = afl((size_t)NW_ * 256);          // r*H scratch
  size_t oU  = afl((size_t)NW_ * 256);          // u scratch
  size_t oFl = afl((size_t)32 * 1536);          // concat(s_fc, e1o, e2o)
  size_t oH1 = afl((size_t)32 * 1024);          // MLP hidden
  size_t oFC = afl((size_t)32 * 512);           // final fc
  size_t nfl = of;

  bf16t* wsb = (bf16t*)(wsf + nfl);
  size_t obf = 0;
  auto abf = [&](size_t n) { size_t r = obf; obf += (n + 63) & ~(size_t)63; return r; };
  size_t pcgkx = abf((size_t)32 * 4 * 512);
  size_t pcgkh = abf((size_t)32 * 8 * 512);
  size_t pcckx = abf((size_t)16 * 4 * 512);
  size_t pcckh = abf((size_t)16 * 8 * 512);
  size_t psgkx = abf((size_t)32 * 20 * 512);
  size_t psgkh = abf((size_t)32 * 8 * 512);
  size_t psckx = abf((size_t)16 * 20 * 512);
  size_t psckh = abf((size_t)16 * 8 * 512);
  size_t psw1  = abf((size_t)64 * 64 * 512);
  size_t psw2  = abf((size_t)32 * 32 * 512);
  size_t pew1  = abf((size_t)64 * 16 * 512);
  size_t pew2  = abf((size_t)32 * 32 * 512);
  size_t pcw1  = abf((size_t)64 * 48 * 512);
  size_t pcw2  = abf((size_t)32 * 32 * 512);

  if (ws_size < nfl * sizeof(float) + obf * sizeof(bf16t)) return;

  auto packW = [&](const float* W, int rowOff, int Ksrc, int KT, int N, size_t dst) {
    size_t tot = (size_t)(N / 16) * KT * 512;
    pack_w_k<<<dim3((unsigned)((tot + 255) / 256)), dim3(256), 0, stream>>>(
        W, rowOff, Ksrc, KT, N, wsb + dst);
  };

  // -------- 1. pack all weights into WMMA B-fragment layout ----------------
  packW(c_gk, 0,   100, 4,  512, pcgkx);
  packW(c_gk, 100, 256, 8,  512, pcgkh);
  packW(c_ck, 0,   100, 4,  256, pcckx);
  packW(c_ck, 100, 256, 8,  256, pcckh);
  packW(s_gk, 0,   612, 20, 512, psgkx);
  packW(s_gk, 612, 256, 8,  512, psgkh);
  packW(s_ck, 0,   612, 20, 256, psckx);
  packW(s_ck, 612, 256, 8,  256, psckh);
  packW(sw1, 0, 2048, 64, 1024, psw1);
  packW(sw2, 0, 1024, 32, 512,  psw2);
  packW(ew1, 0, 512,  16, 1024, pew1);
  packW(ew2, 0, 1024, 32, 512,  pew2);
  packW(cw1, 0, 1536, 48, 1024, pcw1);
  packW(cw2, 0, 1024, 32, 512,  pcw2);

  // -------- 2. gathers + zero-init recurrent states ------------------------
  {
    size_t n = (size_t)NW_ * 512;
    zero_k<<<dim3((unsigned)((n + 255) / 256)), 256, 0, stream>>>(wsf + oCS, n);
  }
  {
    size_t n = (size_t)NSENT_ * 512;
    zero_k<<<dim3((unsigned)((n + 255) / 256)), 256, 0, stream>>>(wsf + oSS, n);
  }
  {
    size_t tot = (size_t)NW_ * LC_ * 128;
    build_xchar_k<<<dim3((unsigned)((tot + 255) / 256)), 256, 0, stream>>>(
        w2c, cemb, wsf + oX);
  }

  // -------- 3. char-level BiGRU scans --------------------------------------
  const int mtC = (NW_ + 15) / 16;  // 313
  for (int dir = 0; dir < 2; ++dir) {
    float* H = wsf + oCS + dir * 256;
    for (int t = 0; t < LC_; ++t) {
      gru_gate_k<<<dim3(mtC, 8), 32, 0, stream>>>(
          wsf + oX, 128, 4, LC_, wlen, nullptr, t, dir, H, 512,
          wsb + pcgkx, wsb + pcgkh, c_gb, wsf + oRH, wsf + oU, NW_);
      gru_update_k<<<dim3(mtC, 4), 32, 0, stream>>>(
          wsf + oX, 128, 4, LC_, wlen, nullptr, t, dir, H, 512,
          wsf + oRH, wsb + pcckx, wsb + pcckh, c_cb, wsf + oU, NW_);
    }
  }

  // -------- 4. build sentence inputs ---------------------------------------
  {
    size_t tot = (size_t)NSENT_ * LW_ * 640;
    build_inputs_k<<<dim3((unsigned)((tot + 255) / 256)), 256, 0, stream>>>(
        s2w, b2s, wsf + oCS, wemb, wsf + oIn);
  }

  // -------- 5. sentence-level BiGRU scans ----------------------------------
  const int mtS = NSENT_ / 16;  // 12
  for (int dir = 0; dir < 2; ++dir) {
    float* H = wsf + oSS + dir * 256;
    for (int t = 0; t < LW_; ++t) {
      gru_gate_k<<<dim3(mtS, 8), 32, 0, stream>>>(
          wsf + oIn, 640, 20, LW_, slen, b2s, t, dir, H, 512,
          wsb + psgkx, wsb + psgkh, s_gb, wsf + oRH, wsf + oU, NSENT_);
      gru_update_k<<<dim3(mtS, 4), 32, 0, stream>>>(
          wsf + oIn, 640, 20, LW_, slen, b2s, t, dir, H, 512,
          wsf + oRH, wsb + psckx, wsb + psckh, s_cb, wsf + oU, NSENT_);
    }
  }

  // -------- 6. MLP head ----------------------------------------------------
  // sent = states[:, :4] -> rows of 2048 contiguous floats, stride 6*512
  mlp_gemm_k<<<dim3(2, 16), 32, 0, stream>>>(wsf + oSS,        3072, wsb + psw1, 64, sb1, wsf + oH1,       1024, 32, 1);
  mlp_gemm_k<<<dim3(2, 8),  32, 0, stream>>>(wsf + oH1,        1024, wsb + psw2, 32, sb2, wsf + oFl,       1536, 32, 1);
  // e1 = states[:,4], e2 = states[:,5]
  mlp_gemm_k<<<dim3(2, 16), 32, 0, stream>>>(wsf + oSS + 2048, 3072, wsb + pew1, 16, eb1, wsf + oH1,       1024, 32, 1);
  mlp_gemm_k<<<dim3(2, 8),  32, 0, stream>>>(wsf + oH1,        1024, wsb + pew2, 32, eb2, wsf + oFl + 512, 1536, 32, 1);
  mlp_gemm_k<<<dim3(2, 16), 32, 0, stream>>>(wsf + oSS + 2560, 3072, wsb + pew1, 16, eb1, wsf + oH1,       1024, 32, 1);
  mlp_gemm_k<<<dim3(2, 8),  32, 0, stream>>>(wsf + oH1,        1024, wsb + pew2, 32, eb2, wsf + oFl + 1024,1536, 32, 1);
  // combine
  mlp_gemm_k<<<dim3(2, 16), 32, 0, stream>>>(wsf + oFl,        1536, wsb + pcw1, 48, cb1, wsf + oH1,       1024, 32, 1);
  mlp_gemm_k<<<dim3(2, 8),  32, 0, stream>>>(wsf + oH1,        1024, wsb + pcw2, 32, cb2, wsf + oFC,        512, 32, 1);

  final_out_k<<<1, 64, 0, stream>>>(wsf + oFC, ow, obv, out);
}
